// GATv2Backbone_12695923327659
// MI455X (gfx1250) — matrix operand order repaired and checked by reading, not compiled, not run on hardware
//
#include <hip/hip_runtime.h>
#include <hip/hip_bf16.h>

// ---------------------------------------------------------------------------
// GATv2 backbone for MI455X (gfx1250, wave32).
//  - Node transforms use v_wmma_f32_16x16x32_f16 (f32 accum), A-fragment
//    shared across Wl/Wr (two accumulators per wave).
//  - Edge phase is bandwidth/atomics bound: b64/b128 gathers,
//    global_atomic_max_u32 (monotonic float keys) for segment-max,
//    global_atomic_add_f32 for exp-sum and the weighted scatter.
// ---------------------------------------------------------------------------

typedef __attribute__((ext_vector_type(16))) _Float16 v16h;
typedef __attribute__((ext_vector_type(8)))  float    v8f;

#define NEG_SLOPE 0.2f

// ---- monotonic order-preserving float<->uint keys (for atomic umax) -------
__device__ __forceinline__ unsigned f32_key(float x) {
  unsigned u = __float_as_uint(x);
  return (u & 0x80000000u) ? ~u : (u | 0x80000000u);
}
__device__ __forceinline__ float key_f32(unsigned k) {
  unsigned u = (k & 0x80000000u) ? (k & 0x7FFFFFFFu) : ~k;
  return __uint_as_float(u);
}

// ---------------------------------------------------------------------------
// Dual GEMM:  XL = X @ Wl, XR = X @ Wr.   X:[N,64] f32, W:[64,FOUT] f32.
// Wave computes a 16x16 output tile for BOTH weight matrices (A reuse).
// Block = 128 threads (4 waves) -> 16 rows x 64 cols.  grid=(N/16, FOUT/64).
// K=64 as two WMMA k-steps of 32 (f16 inputs, f32 accumulate).
// ---------------------------------------------------------------------------
template <int FOUT>
__global__ __launch_bounds__(128)
void gemm_dual(const float* __restrict__ X,
               const float* __restrict__ Wl,
               const float* __restrict__ Wr,
               float* __restrict__ XL,
               float* __restrict__ XR)
{
  const int lane = threadIdx.x & 31;
  const int wave = threadIdx.x >> 5;                 // 0..3
  const int row0 = blockIdx.x * 16;
  const int col0 = (blockIdx.y * 4 + wave) * 16;
  const int r    = lane & 15;
  const int hi   = lane >> 4;                        // 0 or 1 (lane half)

  v8f accL = {};
  v8f accR = {};

  const float* xrow = X + (size_t)(row0 + r) * 64;
  const int    kcol = col0 + r;

#pragma unroll
  for (int ks = 0; ks < 2; ++ks) {
    // A fragment (16x32 f16): lane half 'hi' holds K = hi*16 .. hi*16+15
    const int kb = ks * 32 + hi * 16;
    v16h a;
#pragma unroll
    for (int j = 0; j < 8; ++j) {
      a[j]     = (_Float16)xrow[kb + j];
      a[8 + j] = (_Float16)xrow[kb + 8 + j];
    }
    // B fragment (32x16 f16): lane half holds K = hi*16 .. hi*16+15, col = kcol
    const int kbase = ks * 32 + hi * 16;
    v16h bl, br;
#pragma unroll
    for (int j = 0; j < 16; ++j) {
      bl[j] = (_Float16)Wl[(size_t)(kbase + j) * FOUT + kcol];
      br[j] = (_Float16)Wr[(size_t)(kbase + j) * FOUT + kcol];
    }
    accL = __builtin_amdgcn_wmma_f32_16x16x32_f16(false, a, false, bl,
                                                  (short)0, accL, false, false);
    accR = __builtin_amdgcn_wmma_f32_16x16x32_f16(false, a, false, br,
                                                  (short)0, accR, false, false);
  }

  // D layout: lanes 0-15 -> M = vgpr, lanes 16-31 -> M = vgpr+8; N = lane&15
#pragma unroll
  for (int m = 0; m < 8; ++m) {
    const size_t o = (size_t)(row0 + hi * 8 + m) * FOUT + (col0 + r);
    XL[o] = accL[m];
    XR[o] = accR[m];
  }
}

// ---------------------------------------------------------------------------
// Edge pass 1: per-edge attention logits + per-(dst,head) atomic max.
// One wave per edge; 4 lanes per head (H=8).  C = per-head channels (8 or 64).
// ---------------------------------------------------------------------------
template <int C>
__global__ __launch_bounds__(256)
void edge_logits(const long long* __restrict__ esrc,
                 const long long* __restrict__ edst,
                 int E, int Ep,
                 const float* __restrict__ XL,
                 const float* __restrict__ XR,
                 const float* __restrict__ att,      // [8*C]
                 float* __restrict__ LOGIT,          // [Ep*8]
                 unsigned* __restrict__ MAXU)        // [N*8]
{
  const int lane = threadIdx.x & 31;
  const int wid  = blockIdx.x * (blockDim.x >> 5) + (threadIdx.x >> 5);
  if (wid >= Ep) return;

  int s, d;
  if (wid < E) { s = (int)esrc[wid]; d = (int)edst[wid]; }
  else         { s = d = wid - E; }                  // self loop

  constexpr int PL = C / 4;                          // elems per lane (2 / 16)
  constexpr int FO = 8 * C;
  const int h  = lane >> 2;
  const int c0 = (lane & 3) * PL;

  const float* pl = XL  + (size_t)s * FO + h * C + c0;
  const float* pr = XR  + (size_t)d * FO + h * C + c0;
  const float* pa = att + h * C + c0;

  float acc = 0.f;
  if constexpr (PL == 2) {
    float2 l2 = *(const float2*)pl;
    float2 r2 = *(const float2*)pr;
    float2 a2 = *(const float2*)pa;
    float m0 = l2.x + r2.x, m1 = l2.y + r2.y;
    m0 = m0 > 0.f ? m0 : NEG_SLOPE * m0;
    m1 = m1 > 0.f ? m1 : NEG_SLOPE * m1;
    acc = fmaf(m0, a2.x, m1 * a2.y);
  } else {
#pragma unroll
    for (int q = 0; q < PL / 4; ++q) {
      float4 l4 = ((const float4*)pl)[q];
      float4 r4 = ((const float4*)pr)[q];
      float4 a4 = ((const float4*)pa)[q];
      float m0 = l4.x + r4.x, m1 = l4.y + r4.y;
      float m2 = l4.z + r4.z, m3 = l4.w + r4.w;
      m0 = m0 > 0.f ? m0 : NEG_SLOPE * m0;
      m1 = m1 > 0.f ? m1 : NEG_SLOPE * m1;
      m2 = m2 > 0.f ? m2 : NEG_SLOPE * m2;
      m3 = m3 > 0.f ? m3 : NEG_SLOPE * m3;
      acc = fmaf(m0, a4.x, acc);
      acc = fmaf(m1, a4.y, acc);
      acc = fmaf(m2, a4.z, acc);
      acc = fmaf(m3, a4.w, acc);
    }
  }
  // reduce across the 4 lanes of this head
  acc += __shfl_xor(acc, 1, 32);
  acc += __shfl_xor(acc, 2, 32);
  if ((lane & 3) == 0) {
    LOGIT[(size_t)wid * 8 + h] = acc;
    atomicMax(&MAXU[(size_t)d * 8 + h], f32_key(acc));
  }
}

// ---------------------------------------------------------------------------
// Edge pass 2: e = exp(logit - max[dst]); atomic-add into per-(dst,head) sum.
// Thread per (edge, head).
// ---------------------------------------------------------------------------
__global__ __launch_bounds__(256)
void edge_expsum(const long long* __restrict__ edst,
                 int E, int Ep,
                 const unsigned* __restrict__ MAXU,
                 float* __restrict__ LOGIT,
                 float* __restrict__ SUM)
{
  const int t = blockIdx.x * blockDim.x + threadIdx.x;
  if (t >= Ep * 8) return;
  const int e = t >> 3, h = t & 7;
  const int d = (e < E) ? (int)edst[e] : (e - E);
  const float m = key_f32(MAXU[(size_t)d * 8 + h]);
  const float v = __expf(LOGIT[t] - m);
  LOGIT[t] = v;
  atomicAdd(&SUM[(size_t)d * 8 + h], v);
}

// ---------------------------------------------------------------------------
// Edge pass 3: scatter  ACC[dst] += (e / sum[dst]) * xl[src].
// One wave per edge; global_atomic_add_f32 per element.
// ---------------------------------------------------------------------------
template <int C>
__global__ __launch_bounds__(256)
void edge_scatter(const long long* __restrict__ esrc,
                  const long long* __restrict__ edst,
                  int E, int Ep,
                  const float* __restrict__ XL,
                  const float* __restrict__ LOGIT,
                  const float* __restrict__ SUM,
                  float* __restrict__ ACC)
{
  const int lane = threadIdx.x & 31;
  const int wid  = blockIdx.x * (blockDim.x >> 5) + (threadIdx.x >> 5);
  if (wid >= Ep) return;

  int s, d;
  if (wid < E) { s = (int)esrc[wid]; d = (int)edst[wid]; }
  else         { s = d = wid - E; }

  constexpr int PL = C / 4;
  constexpr int FO = 8 * C;
  const int h  = lane >> 2;
  const int c0 = (lane & 3) * PL;

  const float w = LOGIT[(size_t)wid * 8 + h] /
                  (SUM[(size_t)d * 8 + h] + 1e-16f);

  const float* pl = XL  + (size_t)s * FO + h * C + c0;
  float*       po = ACC + (size_t)d * FO + h * C + c0;

  if constexpr (PL == 2) {
    float2 l2 = *(const float2*)pl;
    atomicAdd(&po[0], w * l2.x);
    atomicAdd(&po[1], w * l2.y);
  } else {
#pragma unroll
    for (int q = 0; q < PL / 4; ++q) {
      float4 l4 = ((const float4*)pl)[q];
      atomicAdd(&po[4 * q + 0], w * l4.x);
      atomicAdd(&po[4 * q + 1], w * l4.y);
      atomicAdd(&po[4 * q + 2], w * l4.z);
      atomicAdd(&po[4 * q + 3], w * l4.w);
    }
  }
}

// ---------------------------------------------------------------------------
// Finalize (layers 0/1): out = elu(acc + bias) [+ residual]   (concat layout)
// ---------------------------------------------------------------------------
__global__ __launch_bounds__(256)
void fin_concat(const float* __restrict__ ACC,
                const float* __restrict__ bias,      // [64]
                const float* __restrict__ resid,     // nullptr or [N*64]
                float* __restrict__ OUT,
                int n64)
{
  const int t = blockIdx.x * blockDim.x + threadIdx.x;
  if (t >= n64) return;
  float v = ACC[t] + bias[t & 63];
  v = v > 0.f ? v : (__expf(v) - 1.f);               // ELU
  if (resid) v += resid[t];
  OUT[t] = v;
}

// ---------------------------------------------------------------------------
// Finalize (layer 2): out[n,c] = mean_h acc[n,h,c] + bias[c]
// ---------------------------------------------------------------------------
__global__ __launch_bounds__(256)
void fin_mean(const float* __restrict__ ACC,         // [N,8,64]
              const float* __restrict__ bias,        // [64]
              float* __restrict__ OUT,               // [N,64]
              int n64)
{
  const int t = blockIdx.x * blockDim.x + threadIdx.x;
  if (t >= n64) return;
  const int n = t >> 6, c = t & 63;
  float s = 0.f;
#pragma unroll
  for (int h = 0; h < 8; ++h) s += ACC[(size_t)n * 512 + h * 64 + c];
  OUT[t] = s * 0.125f + bias[c];
}

// ---------------------------------------------------------------------------
__global__ __launch_bounds__(256)
void zero_f32(float* __restrict__ p, long long n)
{
  long long i = (long long)blockIdx.x * blockDim.x + threadIdx.x;
  const long long stride = (long long)gridDim.x * blockDim.x;
  for (; i < n; i += stride) p[i] = 0.f;
}

// ---------------------------------------------------------------------------
extern "C" void kernel_launch(void* const* d_in, const int* in_sizes, int n_in,
                              void* d_out, int out_size, void* d_ws, size_t ws_size,
                              hipStream_t stream)
{
  const float*     x   = (const float*)d_in[0];
  const long long* ei  = (const long long*)d_in[1];   // int64 [2,E]
  const float* W0l = (const float*)d_in[2];
  const float* W0r = (const float*)d_in[3];
  const float* a0  = (const float*)d_in[4];
  const float* b0  = (const float*)d_in[5];
  const float* W1l = (const float*)d_in[6];
  const float* W1r = (const float*)d_in[7];
  const float* a1  = (const float*)d_in[8];
  const float* b1  = (const float*)d_in[9];
  const float* W2l = (const float*)d_in[10];
  const float* W2r = (const float*)d_in[11];
  const float* a2  = (const float*)d_in[12];
  const float* b2  = (const float*)d_in[13];

  const int N  = in_sizes[0] / 64;
  const int E  = in_sizes[1] / 2;
  const int Ep = E + N;                                // with self loops
  const long long* esrc = ei;
  const long long* edst = ei + E;

  // ---- workspace layout (f32 elems; all offsets 256B-aligned) ----
  float*    XL    = (float*)d_ws;                       // N*512 (max layer)
  float*    XR    = XL   + (size_t)N * 512;             // N*512
  float*    ACC   = XR   + (size_t)N * 512;             // N*512
  float*    LOGIT = ACC  + (size_t)N * 512;             // Ep*8
  float*    SUMf  = LOGIT + (size_t)Ep * 8;             // N*8
  unsigned* MAXU  = (unsigned*)(SUMf + (size_t)N * 8);  // N*8 (monotone keys)
  float*    H1    = (float*)(MAXU + (size_t)N * 8);     // N*64
  float*    H2    = H1   + (size_t)N * 64;              // N*64
  float*    OUT   = (float*)d_out;

  const dim3 gB(256);
  const int  zgrid    = 2048;
  const int  eblocks  = (Ep + 7) / 8;                   // 8 waves/block
  const int  thblocks = (Ep * 8 + 255) / 256;
  const int  nblocks  = (N * 64 + 255) / 256;
  const dim3 gemmG64(N / 16, 1), gemmG512(N / 16, 8), gemmB(128);

  // ================= layer 0 (C=8, FOUT=64) =================
  gemm_dual<64><<<gemmG64, gemmB, 0, stream>>>(x, W0l, W0r, XL, XR);
  zero_f32<<<zgrid, gB, 0, stream>>>(ACC, (long long)N * 64);
  zero_f32<<<zgrid, gB, 0, stream>>>(SUMf, (long long)N * 16);  // SUM + MAXU
  edge_logits<8><<<eblocks, gB, 0, stream>>>(esrc, edst, E, Ep, XL, XR, a0,
                                             LOGIT, MAXU);
  edge_expsum<<<thblocks, gB, 0, stream>>>(edst, E, Ep, MAXU, LOGIT, SUMf);
  edge_scatter<8><<<eblocks, gB, 0, stream>>>(esrc, edst, E, Ep, XL,
                                              LOGIT, SUMf, ACC);
  fin_concat<<<nblocks, gB, 0, stream>>>(ACC, b0, nullptr, H1, N * 64);

  // ================= layer 1 (C=8, FOUT=64) =================
  gemm_dual<64><<<gemmG64, gemmB, 0, stream>>>(H1, W1l, W1r, XL, XR);
  zero_f32<<<zgrid, gB, 0, stream>>>(ACC, (long long)N * 64);
  zero_f32<<<zgrid, gB, 0, stream>>>(SUMf, (long long)N * 16);
  edge_logits<8><<<eblocks, gB, 0, stream>>>(esrc, edst, E, Ep, XL, XR, a1,
                                             LOGIT, MAXU);
  edge_expsum<<<thblocks, gB, 0, stream>>>(edst, E, Ep, MAXU, LOGIT, SUMf);
  edge_scatter<8><<<eblocks, gB, 0, stream>>>(esrc, edst, E, Ep, XL,
                                              LOGIT, SUMf, ACC);
  fin_concat<<<nblocks, gB, 0, stream>>>(ACC, b1, H1, H2, N * 64);

  // ================= layer 2 (C=64, FOUT=512, head-mean) =================
  gemm_dual<512><<<gemmG512, gemmB, 0, stream>>>(H2, W2l, W2r, XL, XR);
  zero_f32<<<zgrid, gB, 0, stream>>>(ACC, (long long)N * 512);
  zero_f32<<<zgrid, gB, 0, stream>>>(SUMf, (long long)N * 16);
  edge_logits<64><<<eblocks, gB, 0, stream>>>(esrc, edst, E, Ep, XL, XR, a2,
                                              LOGIT, MAXU);
  edge_expsum<<<thblocks, gB, 0, stream>>>(edst, E, Ep, MAXU, LOGIT, SUMf);
  edge_scatter<64><<<eblocks, gB, 0, stream>>>(esrc, edst, E, Ep, XL,
                                               LOGIT, SUMf, ACC);
  fin_mean<<<nblocks, gB, 0, stream>>>(ACC, b2, OUT, N * 64);
}
